// LSTMModel_51831665328159
// MI455X (gfx1250) — matrix operand order repaired and checked by reading, not compiled, not run on hardware
//
#include <hip/hip_runtime.h>
#include <hip/hip_bf16.h>
#include <stdint.h>

// Problem dims
#define BB   32
#define SS   512
#define EE   512
#define HH   1024
#define G4H  4096          // 4*H
#define VV   5000
#define VPAD 5056          // 79 * 64

typedef __attribute__((ext_vector_type(16))) __bf16 v16bf;
typedef __attribute__((ext_vector_type(8)))  float  v8f;

union FragBF { v16bf v; uint4 q[2]; };

static __device__ __forceinline__ unsigned short f2bf(float f) {
  unsigned int u = __float_as_uint(f);
  u += 0x7FFFu + ((u >> 16) & 1u);   // round-to-nearest-even
  return (unsigned short)(u >> 16);
}

static __device__ __forceinline__ v8f wmma_bf16(v16bf a, v16bf b, v8f c) {
  return __builtin_amdgcn_wmma_f32_16x16x32_bf16(false, a, false, b, (short)0, c, false, false);
}

// Async DMA: 16 bytes per lane, global -> LDS, tracked by ASYNCcnt (no VGPR data).
// lds_off = LDS byte offset (low 32 bits of the generic pointer to __shared__).
static __device__ __forceinline__ void async_ld_b128(unsigned lds_off, const void* gptr) {
  asm volatile("global_load_async_to_lds_b128 %0, %1, off"
               :: "v"(lds_off), "v"((unsigned long long)(uintptr_t)gptr)
               : "memory");
}
static __device__ __forceinline__ void wait_async0() {
  asm volatile("s_wait_asynccnt 0x0" ::: "memory");
}

// A fragment (16x32, bf16) from row-major global: lane row = m0 + (lane&15)
// elements 0..7 = K[klo..klo+7], 8..15 = K[klo+16..klo+23], klo = (lane<16)?0:8
static __device__ __forceinline__ v16bf load_a_frag(const unsigned short* __restrict__ src,
                                                    int ld, int m0, int kb, int lane) {
  int row = m0 + (lane & 15);
  int klo = (lane < 16) ? 0 : 8;
  const unsigned short* p = src + (long)row * ld + kb + klo;
  FragBF f;
  f.q[0] = *(const uint4*)(p);
  f.q[1] = *(const uint4*)(p + 16);
  return f.v;
}

// B fragment from LDS in packed per-lane layout: chunk c holds 16 bf16 for lane
static __device__ __forceinline__ v16bf load_b_lds(const unsigned short* lds, int chunk) {
  FragBF f;
  f.q[0] = *(const uint4*)(lds + chunk * 16);
  f.q[1] = *(const uint4*)(lds + chunk * 16 + 8);
  return f.v;
}

// ---------------- prep kernels ----------------
__global__ void k_cvt_bf16(const float* __restrict__ src, unsigned short* __restrict__ dst, int n) {
  int i = blockIdx.x * blockDim.x + threadIdx.x;
  if (i < n) dst[i] = f2bf(src[i]);
}

__global__ void k_cvt_wfc_pad(const float* __restrict__ src, unsigned short* __restrict__ dst) {
  int i = blockIdx.x * blockDim.x + threadIdx.x;
  if (i >= VPAD * HH) return;
  int row = i / HH, col = i % HH;
  dst[i] = (row < VV) ? f2bf(src[(long)row * HH + col]) : (unsigned short)0;
}

__global__ void k_zero16(unsigned short* __restrict__ p, int n) {
  int i = blockIdx.x * blockDim.x + threadIdx.x;
  if (i < n) p[i] = 0;
}

__global__ void k_zero32(unsigned int* __restrict__ p, int n) {
  int i = blockIdx.x * blockDim.x + threadIdx.x;
  if (i < n) p[i] = 0u;
}

// ---------------- phase 1: xg = emb[x] @ W_ih^T + b_ih + b_hh ----------------
// Block: 256 rows x 64 cols. 8 waves, each 2 M-tiles x 4 N-tiles.
// B tile double-buffered in LDS (2 x 64col x 32k bf16 = 8KB), fragment layout,
// staged with async global->LDS DMA overlapped with the WMMAs of the current k-step.
__global__ void k_gemm_xg(const int* __restrict__ x,
                          const unsigned short* __restrict__ emb_bf,
                          const unsigned short* __restrict__ wih_bf,
                          const float* __restrict__ b_ih,
                          const float* __restrict__ b_hh,
                          float* __restrict__ xg) {
  __shared__ unsigned short ldsb[2 * 4 * 32 * 16];   // [buf][j][lane][16]
  int tid  = threadIdx.x;
  int w    = tid >> 5;
  int lane = tid & 31;
  int lo   = lane & 15;
  int hi8  = (lane < 16) ? 0 : 8;

  int mb = blockIdx.x >> 6;        // 0..63
  int ng = blockIdx.x & 63;        // 0..63
  int n0 = ng * 64;
  int m0 = mb * 256 + w * 32;      // this wave: rows m0..m0+31

  // token gather for the two A tiles
  const unsigned short* aptr0 = emb_bf + (long)x[m0 + lo] * EE;
  const unsigned short* aptr1 = emb_bf + (long)x[m0 + 16 + lo] * EE;
  int klo_a = (lane < 16) ? 0 : 8;

  // staging map: tid -> (j, L, half); each thread DMAs 16B per k-step
  int sj = tid >> 6;               // 0..3
  int sL = (tid >> 1) & 31;        // 0..31
  int sh = tid & 1;                // 0..1
  const unsigned short* sgsrc =
      wih_bf + (long)(n0 + sj * 16 + (sL & 15)) * EE + ((sL < 16) ? 0 : 8) + sh * 16;
  unsigned lds_off0 = (unsigned)(uintptr_t)&ldsb[((sj) * 32 + sL) * 16 + sh * 8];     // buf0
  unsigned lds_off1 = (unsigned)(uintptr_t)&ldsb[((4 + sj) * 32 + sL) * 16 + sh * 8]; // buf1

  v8f acc[4][2];
#pragma unroll
  for (int j = 0; j < 4; ++j) {
    float bias = b_ih[n0 + j * 16 + lo] + b_hh[n0 + j * 16 + lo];
#pragma unroll
    for (int mt = 0; mt < 2; ++mt)
#pragma unroll
      for (int r = 0; r < 8; ++r) acc[j][mt][r] = bias;
  }

  // prologue: DMA k-block 0 into buf0
  async_ld_b128(lds_off0, sgsrc);
  wait_async0();
  __syncthreads();

  const int KT = EE / 32;
  for (int kt = 0; kt < KT; ++kt) {
    int kb = kt * 32;
    int p  = kt & 1;
    if (kt + 1 < KT)
      async_ld_b128(p ? lds_off0 : lds_off1, sgsrc + kb + 32);
    FragBF a0, a1;
    a0.q[0] = *(const uint4*)(aptr0 + kb + klo_a);
    a0.q[1] = *(const uint4*)(aptr0 + kb + klo_a + 16);
    a1.q[0] = *(const uint4*)(aptr1 + kb + klo_a);
    a1.q[1] = *(const uint4*)(aptr1 + kb + klo_a + 16);
#pragma unroll
    for (int j = 0; j < 4; ++j) {
      v16bf bf = load_b_lds(ldsb, (p * 4 + j) * 32 + lane);
      acc[j][0] = wmma_bf16(a0.v, bf, acc[j][0]);
      acc[j][1] = wmma_bf16(a1.v, bf, acc[j][1]);
    }
    wait_async0();
    __syncthreads();
  }

#pragma unroll
  for (int j = 0; j < 4; ++j)
#pragma unroll
    for (int mt = 0; mt < 2; ++mt)
#pragma unroll
      for (int r = 0; r < 8; ++r)
        xg[(long)(m0 + mt * 16 + r + hi8) * G4H + (n0 + j * 16 + lo)] = acc[j][mt][r];
}

// ---------------- phase 2: persistent LSTM (all 512 steps in one kernel) ----------------
// 32 blocks x 64 threads (2 waves). Each wave owns 16 hidden units (4 gates, 32 batch rows).
// Its 64x1024 W_hh slice lives in LDS in WMMA-B fragment layout (128KB/wave, 256KB/block),
// filled once via async DMA. Steps separated by a grid-wide barrier (monotonic counter).
__global__ void k_lstm_persistent(const float* __restrict__ xg,
                                  const unsigned short* __restrict__ whh_bf,
                                  unsigned short* __restrict__ h0,
                                  unsigned short* __restrict__ h1,
                                  float* __restrict__ cbuf,
                                  unsigned short* __restrict__ hall,
                                  unsigned int* __restrict__ barcnt) {
  extern __shared__ unsigned short lds[];   // 2*32*4*32 chunks * 16 bf16 = 256KB
  int tid  = threadIdx.x;
  int w    = tid >> 5;
  int lane = tid & 31;
  int lo   = lane & 15;
  int hi8  = (lane < 16) ? 0 : 8;
  int blk  = blockIdx.x;                    // 0..31
  int ub   = blk * 32 + w * 16;             // this wave's unit base
  int u    = ub + lo;

  // --- one-time async DMA fill of W_hh slice, fragment layout ---
  // chunk c = ((w2*32 + kt)*4 + g)*32 + L  -> 16 bf16 for lane L
  for (int c = tid; c < 8192; c += 64) {
    int L  = c & 31;
    int g  = (c >> 5) & 3;
    int kt = (c >> 7) & 31;
    int w2 = (c >> 12) & 1;
    int row  = g * HH + blk * 32 + w2 * 16 + (L & 15);
    int koff = kt * 32 + ((L < 16) ? 0 : 8);
    const unsigned short* src = whh_bf + (long)row * HH + koff;
    unsigned dst = (unsigned)(uintptr_t)(lds + c * 16);
    async_ld_b128(dst, src);
    async_ld_b128(dst + 16, src + 16);
  }
  wait_async0();
  __syncthreads();

  for (int t = 0; t < SS; ++t) {
    const unsigned short* hp = (t & 1) ? h1 : h0;
    unsigned short*       hn = (t & 1) ? h0 : h1;

    v8f acc[4][2];
#pragma unroll
    for (int g = 0; g < 4; ++g)
#pragma unroll
      for (int mt = 0; mt < 2; ++mt)
#pragma unroll
        for (int r = 0; r < 8; ++r) {
          int b = mt * 16 + r + hi8;
          acc[g][mt][r] = xg[(long)(b * SS + t) * G4H + g * HH + u];
        }

    for (int kt = 0; kt < HH / 32; ++kt) {
      int kb = kt * 32;
      v16bf a0 = load_a_frag(hp, HH, 0,  kb, lane);
      v16bf a1 = load_a_frag(hp, HH, 16, kb, lane);
#pragma unroll
      for (int g = 0; g < 4; ++g) {
        v16bf bf = load_b_lds(lds, ((w * 32 + kt) * 4 + g) * 32 + lane);
        acc[g][0] = wmma_bf16(a0, bf, acc[g][0]);
        acc[g][1] = wmma_bf16(a1, bf, acc[g][1]);
      }
    }

#pragma unroll
    for (int mt = 0; mt < 2; ++mt)
#pragma unroll
      for (int r = 0; r < 8; ++r) {
        int b = mt * 16 + r + hi8;
        float gi = acc[0][mt][r];
        float gf = acc[1][mt][r];
        float gg = acc[2][mt][r];
        float go = acc[3][mt][r];
        gi = 1.0f / (1.0f + __expf(-gi));
        gf = 1.0f / (1.0f + __expf(-gf));
        gg = tanhf(gg);
        go = 1.0f / (1.0f + __expf(-go));
        float c = gf * cbuf[b * HH + u] + gi * gg;
        cbuf[b * HH + u] = c;
        float h = go * tanhf(c);
        unsigned short hb = f2bf(h);
        hn[b * HH + u] = hb;
        hall[(long)(b * SS + t) * HH + u] = hb;
      }

    // grid barrier: release fence, arrive, spin, acquire fence (all waves fence)
    __threadfence();
    __syncthreads();
    if (tid == 0) {
      atomicAdd(barcnt, 1u);
      unsigned int target = 32u * (unsigned int)(t + 1);
      while (*(volatile unsigned int*)barcnt < target) __builtin_amdgcn_s_sleep(2);
    }
    __syncthreads();
    __threadfence();
  }
}

// ---------------- phase 3: logits = h_all @ W_fc^T + b_fc ----------------
__global__ void k_gemm_fc(const unsigned short* __restrict__ hall,
                          const unsigned short* __restrict__ wfc_bf,
                          const float* __restrict__ b_fc,
                          float* __restrict__ out) {
  __shared__ unsigned short ldsb[2 * 4 * 32 * 16];   // 8KB
  int tid  = threadIdx.x;
  int w    = tid >> 5;
  int lane = tid & 31;
  int lo   = lane & 15;
  int hi8  = (lane < 16) ? 0 : 8;

  int mb = blockIdx.x / 79;        // 0..63
  int ng = blockIdx.x % 79;        // 0..78
  int n0 = ng * 64;
  int m0 = mb * 256 + w * 32;

  int sj = tid >> 6;
  int sL = (tid >> 1) & 31;
  int sh = tid & 1;
  const unsigned short* sgsrc =
      wfc_bf + (long)(n0 + sj * 16 + (sL & 15)) * HH + ((sL < 16) ? 0 : 8) + sh * 16;
  unsigned lds_off0 = (unsigned)(uintptr_t)&ldsb[((sj) * 32 + sL) * 16 + sh * 8];
  unsigned lds_off1 = (unsigned)(uintptr_t)&ldsb[((4 + sj) * 32 + sL) * 16 + sh * 8];

  v8f acc[4][2];
#pragma unroll
  for (int j = 0; j < 4; ++j) {
    int n = n0 + j * 16 + lo;
    float bias = (n < VV) ? b_fc[n] : 0.0f;
#pragma unroll
    for (int mt = 0; mt < 2; ++mt)
#pragma unroll
      for (int r = 0; r < 8; ++r) acc[j][mt][r] = bias;
  }

  async_ld_b128(lds_off0, sgsrc);
  wait_async0();
  __syncthreads();

  const int KT = HH / 32;
  for (int kt = 0; kt < KT; ++kt) {
    int kb = kt * 32;
    int p  = kt & 1;
    if (kt + 1 < KT)
      async_ld_b128(p ? lds_off0 : lds_off1, sgsrc + kb + 32);
    v16bf a0 = load_a_frag(hall, HH, m0,      kb, lane);
    v16bf a1 = load_a_frag(hall, HH, m0 + 16, kb, lane);
#pragma unroll
    for (int j = 0; j < 4; ++j) {
      v16bf bf = load_b_lds(ldsb, (p * 4 + j) * 32 + lane);
      acc[j][0] = wmma_bf16(a0, bf, acc[j][0]);
      acc[j][1] = wmma_bf16(a1, bf, acc[j][1]);
    }
    wait_async0();
    __syncthreads();
  }

#pragma unroll
  for (int j = 0; j < 4; ++j) {
    int n = n0 + j * 16 + lo;
    if (n < VV) {
#pragma unroll
      for (int mt = 0; mt < 2; ++mt)
#pragma unroll
        for (int r = 0; r < 8; ++r)
          out[(long)(m0 + mt * 16 + r + hi8) * VV + n] = acc[j][mt][r];
    }
  }
}

// ---------------- launch ----------------
extern "C" void kernel_launch(void* const* d_in, const int* in_sizes, int n_in,
                              void* d_out, int out_size, void* d_ws, size_t ws_size,
                              hipStream_t stream) {
  (void)in_sizes; (void)n_in; (void)out_size; (void)ws_size;
  const int*   x    = (const int*)  d_in[0];
  const float* emb  = (const float*)d_in[1];
  const float* W_ih = (const float*)d_in[2];
  const float* W_hh = (const float*)d_in[3];
  const float* b_ih = (const float*)d_in[4];
  const float* b_hh = (const float*)d_in[5];
  const float* W_fc = (const float*)d_in[6];
  const float* b_fc = (const float*)d_in[7];
  float* out = (float*)d_out;

  char* ws = (char*)d_ws;
  size_t off = 0;
  auto alloc = [&](size_t bytes) -> char* {
    char* p = ws + off;
    off = (off + bytes + 255) & ~(size_t)255;
    return p;
  };
  unsigned short* emb_bf = (unsigned short*)alloc((size_t)VV * EE * 2);
  unsigned short* wih_bf = (unsigned short*)alloc((size_t)G4H * EE * 2);
  unsigned short* whh_bf = (unsigned short*)alloc((size_t)G4H * HH * 2);
  unsigned short* wfc_bf = (unsigned short*)alloc((size_t)VPAD * HH * 2);
  float*          xg     = (float*)         alloc((size_t)BB * SS * G4H * 4);
  unsigned short* hall   = (unsigned short*)alloc((size_t)BB * SS * HH * 2);
  unsigned short* hbuf0  = (unsigned short*)alloc((size_t)BB * HH * 2);
  unsigned short* hbuf1  = (unsigned short*)alloc((size_t)BB * HH * 2);
  float*          cbuf   = (float*)         alloc((size_t)BB * HH * 4);
  unsigned int*   barcnt = (unsigned int*)  alloc(256);

  // --- prep: bf16 conversions + state zeroing (also resets barrier counter) ---
  {
    int n;
    n = VV * EE;   k_cvt_bf16<<<(n + 255) / 256, 256, 0, stream>>>(emb,  emb_bf, n);
    n = G4H * EE;  k_cvt_bf16<<<(n + 255) / 256, 256, 0, stream>>>(W_ih, wih_bf, n);
    n = G4H * HH;  k_cvt_bf16<<<(n + 255) / 256, 256, 0, stream>>>(W_hh, whh_bf, n);
    n = VPAD * HH; k_cvt_wfc_pad<<<(n + 255) / 256, 256, 0, stream>>>(W_fc, wfc_bf);
    n = BB * HH;   k_zero16<<<(n + 255) / 256, 256, 0, stream>>>(hbuf0, n);
    n = BB * HH;   k_zero16<<<(n + 255) / 256, 256, 0, stream>>>(hbuf1, n);
    n = BB * HH;   k_zero32<<<(n + 255) / 256, 256, 0, stream>>>((unsigned int*)cbuf, n);
    k_zero32<<<1, 64, 0, stream>>>(barcnt, 64);
  }

  // --- phase 1: gate-input GEMM, async-staged B, 64 mblocks x 64 ngroups ---
  k_gemm_xg<<<64 * 64, 256, 0, stream>>>(x, emb_bf, wih_bf, b_ih, b_hh, xg);

  // --- phase 2: persistent LSTM, W_hh resident in LDS (256KB dynamic) ---
  k_lstm_persistent<<<32, 64, 256 * 1024, stream>>>(xg, whh_bf, hbuf0, hbuf1,
                                                    cbuf, hall, barcnt);

  // --- phase 3: vocab projection GEMM, 64 mblocks x 79 ngroups ---
  k_gemm_fc<<<64 * 79, 256, 0, stream>>>(hall, wfc_bf, b_fc, out);
}
